// MultiHeadSelfAttention_24764781429511
// MI455X (gfx1250) — matrix operand order repaired and checked by reading, not compiled, not run on hardware
//
#include <hip/hip_runtime.h>
#include <stdint.h>

typedef __bf16 bf16;
typedef __attribute__((ext_vector_type(8)))  bf16  v8bf;
typedef __attribute__((ext_vector_type(16))) bf16  v16bf;
typedef __attribute__((ext_vector_type(8)))  float v8f;

union V16 { v16bf v; v8bf h[2]; };

#define D_EMB 1024
#define SEQ   2048
#define NH    16
#define DK    64
// total rows M = B*S = 4096

// --- CDNA5 async global->LDS staging (ASYNCcnt), ISA 15.18.3 op 98 ---------
#define ASYNC_STAGE_B128(ldsaddr, gaddr)                                     \
  asm volatile("global_load_async_to_lds_b128 %0, %1, off"                   \
               :: "v"(ldsaddr), "v"(gaddr) : "memory")
#define WAIT_ASYNC_0() asm volatile("s_wait_asynccnt 0x0" ::: "memory")
#define WAIT_ASYNC_1() asm volatile("s_wait_asynccnt 0x1" ::: "memory")

// generic LDS pointer: low 32 bits hold the LDS byte offset (aperture form)
__device__ __forceinline__ unsigned lds_off(const void* p) {
  return (unsigned)(uintptr_t)p;
}

// ---------------------------------------------------------------------------
// fp32 -> bf16 convert, 8 elements / thread
// ---------------------------------------------------------------------------
__global__ __launch_bounds__(256) void cvt_f32_bf16(const float* __restrict__ src,
                                                    bf16* __restrict__ dst, int n8) {
  int i = blockIdx.x * blockDim.x + threadIdx.x;
  if (i >= n8) return;
  const float4* s4 = reinterpret_cast<const float4*>(src);
  float4 a = s4[2 * i], b = s4[2 * i + 1];
  v8bf o;
  o[0] = (bf16)a.x; o[1] = (bf16)a.y; o[2] = (bf16)a.z; o[3] = (bf16)a.w;
  o[4] = (bf16)b.x; o[5] = (bf16)b.y; o[6] = (bf16)b.z; o[7] = (bf16)b.w;
  reinterpret_cast<v8bf*>(dst)[i] = o;
}

// ---------------------------------------------------------------------------
// C[m,n] = sum_k A[m,k] * W[n,k]   (y = A @ W^T), M=4096, N=K=1024
// 256 thr (8 waves); block tile 256(M) x 64(N); wave tile 32 x 64.
// W panel async-staged into double-buffered LDS (each frag feeds 2 WMMAs);
// A fragments register-pipelined, issued after the top barrier.
// ---------------------------------------------------------------------------
__global__ __launch_bounds__(256) void gemm_xwt(const bf16* __restrict__ A,
                                                const bf16* __restrict__ W,
                                                bf16* __restrict__ outB,
                                                float* __restrict__ outF) {
  __shared__ __attribute__((aligned(16))) bf16 Bs[2][64 * 32];

  const int tid  = threadIdx.x;
  const int wave = tid >> 5, lane = tid & 31;
  const int lm = lane & 15, lh = lane >> 4;
  const int n0 = blockIdx.x * 64;
  const int m0 = blockIdx.y * 256 + wave * 32;

  const v8f z = {0.f, 0.f, 0.f, 0.f, 0.f, 0.f, 0.f, 0.f};
  v8f acc[2][4] = {{z, z, z, z}, {z, z, z, z}};

  const size_t arow0      = (size_t)(m0 + lm) * D_EMB;
  const size_t arow1      = (size_t)(m0 + 16 + lm) * D_EMB;
  const size_t wrow_stage = (size_t)(n0 + (tid >> 2)) * D_EMB + (tid & 3) * 8;
  const int    srow_stage = (tid >> 2) * 32 + (tid & 3) * 8;

  const unsigned ldsB0 = lds_off(&Bs[0][srow_stage]);
  const unsigned ldsB1 = lds_off(&Bs[1][srow_stage]);
  const uint64_t gW    = (uint64_t)(uintptr_t)&W[wrow_stage];

  // prologue: stage k-step 0, preload A fragments for k-step 0
  ASYNC_STAGE_B128(ldsB0, gW);
  V16 a_cur[2], a_nxt[2];
  a_cur[0].h[0] = *reinterpret_cast<const v8bf*>(&A[arow0 + lh * 8]);
  a_cur[0].h[1] = *reinterpret_cast<const v8bf*>(&A[arow0 + 16 + lh * 8]);
  a_cur[1].h[0] = *reinterpret_cast<const v8bf*>(&A[arow1 + lh * 8]);
  a_cur[1].h[1] = *reinterpret_cast<const v8bf*>(&A[arow1 + 16 + lh * 8]);

  for (int k0 = 0; k0 < D_EMB; k0 += 32) {
    const int  cur  = (k0 >> 5) & 1;
    const bool last = (k0 + 32 >= D_EMB);
    if (!last) {
      ASYNC_STAGE_B128(cur ? ldsB0 : ldsB1, gW + (uint64_t)(k0 + 32) * 2);
      WAIT_ASYNC_1();                 // async completes in-order: cur tile done
    } else {
      WAIT_ASYNC_0();
    }
    __syncthreads();                  // cur tile visible to all waves

    if (!last) {                      // issue next A frags; consumed next iter
      a_nxt[0].h[0] = *reinterpret_cast<const v8bf*>(&A[arow0 + k0 + 32 + lh * 8]);
      a_nxt[0].h[1] = *reinterpret_cast<const v8bf*>(&A[arow0 + k0 + 48 + lh * 8]);
      a_nxt[1].h[0] = *reinterpret_cast<const v8bf*>(&A[arow1 + k0 + 32 + lh * 8]);
      a_nxt[1].h[1] = *reinterpret_cast<const v8bf*>(&A[arow1 + k0 + 48 + lh * 8]);
    }

    const bf16* bs = &Bs[cur][0];
#pragma unroll
    for (int nt = 0; nt < 4; ++nt) {
      V16 b;
      const bf16* bp = bs + (nt * 16 + lm) * 32;
      b.h[0] = *reinterpret_cast<const v8bf*>(bp + lh * 8);
      b.h[1] = *reinterpret_cast<const v8bf*>(bp + 16 + lh * 8);
      acc[0][nt] = __builtin_amdgcn_wmma_f32_16x16x32_bf16(
          false, a_cur[0].v, false, b.v, (short)0, acc[0][nt], false, false);
      acc[1][nt] = __builtin_amdgcn_wmma_f32_16x16x32_bf16(
          false, a_cur[1].v, false, b.v, (short)0, acc[1][nt], false, false);
    }
    __syncthreads();                  // all reads of cur done before overwrite
    a_cur[0] = a_nxt[0];
    a_cur[1] = a_nxt[1];
  }

#pragma unroll
  for (int mt = 0; mt < 2; ++mt)
#pragma unroll
    for (int nt = 0; nt < 4; ++nt)
#pragma unroll
      for (int j = 0; j < 8; ++j) {
        const size_t row = (size_t)(m0 + mt * 16 + j + 8 * lh);
        const size_t col = (size_t)(n0 + nt * 16 + lm);
        if (outF) outF[row * D_EMB + col] = acc[mt][nt][j];
        else      outB[row * D_EMB + col] = (bf16)acc[mt][nt][j];
      }
}

// ---------------------------------------------------------------------------
// Fused attention: per block one (b,h) and a 256-row q block; each wave owns
// two 16-row q tiles so every staged K/V fragment feeds 2 WMMAs.
// Single-pass softmax (clip(+-5) bounds exp -> no running max needed).
// K async-staged double-buffered; V transposed through VGPRs, double-buffered.
// ---------------------------------------------------------------------------
__global__ __launch_bounds__(256) void attn_fused(const bf16* __restrict__ Qb,
                                                  const bf16* __restrict__ Kb,
                                                  const bf16* __restrict__ Vb,
                                                  bf16* __restrict__ Cb) {
  __shared__ __attribute__((aligned(16))) bf16 Ks[2][32 * 64];      // [k][d]
  __shared__ __attribute__((aligned(16))) bf16 Vt[2][64 * 32];      // [d][k]
  __shared__ __attribute__((aligned(16))) bf16 Ps[8][2][16 * 32];   // per-wave P

  const int tid  = threadIdx.x;
  const int wave = tid >> 5, lane = tid & 31;
  const int lm = lane & 15, lh = lane >> 4;
  const int bh = blockIdx.y;
  const int bb = bh >> 4, h = bh & 15;
  const int sbase = bb * SEQ;
  const int q0    = blockIdx.x * 256 + wave * 32;
  const int hcol  = h * DK;

  // Preload Q A-fragments: 2 q-tiles x two 16x32 d-chunks
  V16 qa[2][2];
#pragma unroll
  for (int qt = 0; qt < 2; ++qt) {
    const bf16* qrow = &Qb[(size_t)(sbase + q0 + qt * 16 + lm) * D_EMB + hcol];
#pragma unroll
    for (int c = 0; c < 2; ++c) {
      qa[qt][c].h[0] = *reinterpret_cast<const v8bf*>(qrow + c * 32 + lh * 8);
      qa[qt][c].h[1] = *reinterpret_cast<const v8bf*>(qrow + c * 32 + 16 + lh * 8);
    }
  }

  const v8f z = {0.f, 0.f, 0.f, 0.f, 0.f, 0.f, 0.f, 0.f};
  v8f octx[2][4] = {{z, z, z, z}, {z, z, z, z}};
  float dsum[2][8] = {{0.f, 0.f, 0.f, 0.f, 0.f, 0.f, 0.f, 0.f},
                      {0.f, 0.f, 0.f, 0.f, 0.f, 0.f, 0.f, 0.f}};

  const int kl   = tid >> 3;        // 0..31 : k row staged by this thread
  const int dseg = (tid & 7) * 8;   // 0..56 : d sub-segment

  const unsigned ldsK0 = lds_off(&Ks[0][kl * 64 + dseg]);
  const unsigned ldsK1 = lds_off(&Ks[1][kl * 64 + dseg]);
  const bf16* gK = &Kb[(size_t)(sbase + kl) * D_EMB + hcol + dseg];
  const bf16* gV = &Vb[(size_t)(sbase + kl) * D_EMB + hcol + dseg];

  // prologue: stage tile 0 (K async, V via VGPR transpose)
  ASYNC_STAGE_B128(ldsK0, (uint64_t)(uintptr_t)gK);
  {
    v8bf vv = *reinterpret_cast<const v8bf*>(gV);
#pragma unroll
    for (int i = 0; i < 8; ++i) Vt[0][(dseg + i) * 32 + kl] = vv[i];
  }

  for (int kb = 0; kb < SEQ; kb += 32) {
    const int  cur  = (kb >> 5) & 1;
    const bool last = (kb + 32 >= SEQ);
    v8bf vnxt;
    if (!last) {
      ASYNC_STAGE_B128(cur ? ldsK0 : ldsK1,
                       (uint64_t)(uintptr_t)(gK + (size_t)(kb + 32) * D_EMB));
      WAIT_ASYNC_1();
    } else {
      WAIT_ASYNC_0();
    }
    __syncthreads();                  // Ks[cur] + Vt[cur] ready

    if (!last)                        // next V tile; consumed after bottom sync
      vnxt = *reinterpret_cast<const v8bf*>(gV + (size_t)(kb + 32) * D_EMB);

    // scores = Q K^T (2 x 16 q x 32 k), then p = exp(clip(s/8))
#pragma unroll
    for (int nt = 0; nt < 2; ++nt) {
      V16 bK[2];
#pragma unroll
      for (int c = 0; c < 2; ++c) {
        const bf16* kp = &Ks[cur][(nt * 16 + lm) * 64 + c * 32];
        bK[c].h[0] = *reinterpret_cast<const v8bf*>(kp + lh * 8);
        bK[c].h[1] = *reinterpret_cast<const v8bf*>(kp + 16 + lh * 8);
      }
#pragma unroll
      for (int qt = 0; qt < 2; ++qt) {
        v8f s = z;
        s = __builtin_amdgcn_wmma_f32_16x16x32_bf16(
            false, qa[qt][0].v, false, bK[0].v, (short)0, s, false, false);
        s = __builtin_amdgcn_wmma_f32_16x16x32_bf16(
            false, qa[qt][1].v, false, bK[1].v, (short)0, s, false, false);
#pragma unroll
        for (int j = 0; j < 8; ++j) {
          float x = s[j] * 0.125f;                 // 1/sqrt(64)
          x = fminf(5.0f, fmaxf(-5.0f, x));        // reference clip
          float p = __expf(x);                     // bounded: no max-subtract
          dsum[qt][j] += p;
          Ps[wave][qt][(j + 8 * lh) * 32 + nt * 16 + lm] = (bf16)p;
        }
      }
    }

    // reload P in A-layout (LDS ops are in-order within a wave)
    V16 pa[2];
#pragma unroll
    for (int qt = 0; qt < 2; ++qt) {
      pa[qt].h[0] = *reinterpret_cast<const v8bf*>(&Ps[wave][qt][lm * 32 + lh * 8]);
      pa[qt].h[1] = *reinterpret_cast<const v8bf*>(&Ps[wave][qt][lm * 32 + 16 + lh * 8]);
    }

    // octx += P @ V  (each bV fragment feeds both q tiles)
#pragma unroll
    for (int dt = 0; dt < 4; ++dt) {
      V16 bV;
      const bf16* vp = &Vt[cur][(dt * 16 + lm) * 32];
      bV.h[0] = *reinterpret_cast<const v8bf*>(vp + lh * 8);
      bV.h[1] = *reinterpret_cast<const v8bf*>(vp + 16 + lh * 8);
#pragma unroll
      for (int qt = 0; qt < 2; ++qt)
        octx[qt][dt] = __builtin_amdgcn_wmma_f32_16x16x32_bf16(
            false, pa[qt].v, false, bV.v, (short)0, octx[qt][dt], false, false);
    }
    __syncthreads();                  // all reads of cur done

    if (!last) {                      // scatter next V tile (transposed)
#pragma unroll
      for (int i = 0; i < 8; ++i) Vt[cur ^ 1][(dseg + i) * 32 + kl] = vnxt[i];
    }
  }

  // denominator: reduce across the 16 lanes of each half-wave
#pragma unroll
  for (int qt = 0; qt < 2; ++qt)
#pragma unroll
    for (int j = 0; j < 8; ++j) {
#pragma unroll
      for (int off = 1; off < 16; off <<= 1)
        dsum[qt][j] += __shfl_xor(dsum[qt][j], off, 16);
      dsum[qt][j] = 1.0f / dsum[qt][j];
    }

#pragma unroll
  for (int qt = 0; qt < 2; ++qt)
#pragma unroll
    for (int dt = 0; dt < 4; ++dt)
#pragma unroll
      for (int j = 0; j < 8; ++j) {
        const size_t row = (size_t)(sbase + q0 + qt * 16 + j + 8 * lh);
        Cb[row * D_EMB + hcol + dt * 16 + lm] =
            (bf16)(octx[qt][dt][j] * dsum[qt][j]);
      }
}

// ---------------------------------------------------------------------------
extern "C" void kernel_launch(void* const* d_in, const int* in_sizes, int n_in,
                              void* d_out, int out_size, void* d_ws, size_t ws_size,
                              hipStream_t stream) {
  const float* nin = (const float*)d_in[0];
  const float* Wq  = (const float*)d_in[1];
  const float* Wk  = (const float*)d_in[2];
  const float* Wv  = (const float*)d_in[3];
  const float* Wo  = (const float*)d_in[4];

  const size_t MX = (size_t)2 * SEQ * D_EMB;  // 4M elements (activations)
  const size_t MW = (size_t)D_EMB * D_EMB;    // 1M elements (one weight)

  bf16* Xb  = (bf16*)d_ws;     // bf16 scratch layout (48 MB total)
  bf16* Wqb = Xb  + MX;
  bf16* Wkb = Wqb + MW;
  bf16* Wvb = Wkb + MW;
  bf16* Wvb2= Wvb;             // (alias, silence unused warnings if any)
  bf16* Wob = Wvb + MW;
  bf16* Qb  = Wob + MW;
  bf16* Kb  = Qb  + MX;
  bf16* Vb  = Kb  + MX;
  bf16* Cb  = Vb  + MX;        // attention context
  (void)Wvb2;

  cvt_f32_bf16<<<dim3((unsigned)(MX / 8 / 256)), 256, 0, stream>>>(nin, Xb, (int)(MX / 8));
  cvt_f32_bf16<<<dim3((unsigned)(MW / 8 / 256)), 256, 0, stream>>>(Wq, Wqb, (int)(MW / 8));
  cvt_f32_bf16<<<dim3((unsigned)(MW / 8 / 256)), 256, 0, stream>>>(Wk, Wkb, (int)(MW / 8));
  cvt_f32_bf16<<<dim3((unsigned)(MW / 8 / 256)), 256, 0, stream>>>(Wv, Wvb, (int)(MW / 8));
  cvt_f32_bf16<<<dim3((unsigned)(MW / 8 / 256)), 256, 0, stream>>>(Wo, Wob, (int)(MW / 8));

  dim3 gg(D_EMB / 64, (2 * SEQ) / 256);  // (16, 16)
  gemm_xwt<<<gg, 256, 0, stream>>>(Xb, Wqb, Qb, nullptr);
  gemm_xwt<<<gg, 256, 0, stream>>>(Xb, Wkb, Kb, nullptr);
  gemm_xwt<<<gg, 256, 0, stream>>>(Xb, Wvb, Vb, nullptr);

  attn_fused<<<dim3(SEQ / 256, 2 * NH), 256, 0, stream>>>(Qb, Kb, Vb, Cb);

  gemm_xwt<<<gg, 256, 0, stream>>>(Cb, Wob, nullptr, (float*)d_out);
}
